// DPASSMBlock_16681652978422
// MI455X (gfx1250) — matrix operand order repaired and checked by reading, not compiled, not run on hardware
//
#include <hip/hip_runtime.h>
#include <math.h>

// ---------------- problem constants ----------------
#define BSZ 2
#define TSEQ 2048
#define DMODEL 1024
#define NHEAD 16
#define DHEAD 64
#define WINSZ 128
#define NSSM 64
#define DFFN 4096
#define MROWS (BSZ * TSEQ)   // 4096

typedef __bf16 bf16;
typedef __bf16 v16bf __attribute__((ext_vector_type(16)));
typedef float  v8f   __attribute__((ext_vector_type(8)));

#define WMMA_BF16(a, b, c) \
  __builtin_amdgcn_wmma_f32_16x16x32_bf16(false, (a), false, (b), (short)0, (c), false, false)

typedef int v4i __attribute__((vector_size(4 * sizeof(int))));
typedef __attribute__((address_space(1))) v4i gv4i;   // global
typedef __attribute__((address_space(3))) v4i lv4i;   // LDS

// Async global->LDS 16B copy (CDNA5 path, ASYNCcnt). Falls back to a
// synchronous copy if this toolchain lacks the builtin (also guards host pass).
__device__ __forceinline__ void async_b128(void* lds, const void* g) {
#if __has_builtin(__builtin_amdgcn_global_load_async_to_lds_b128)
  __builtin_amdgcn_global_load_async_to_lds_b128((gv4i*)g, (lv4i*)lds, 0, 0);
#else
  *(uint4*)lds = *(const uint4*)g;
#endif
}
__device__ __forceinline__ void async_wait0() {
#if __has_builtin(__builtin_amdgcn_s_wait_asynccnt)
  __builtin_amdgcn_s_wait_asynccnt(0);
#endif
}

// A-matrix 16x32 bf16 fragment (ISA 7.12.2): lane m = lane&15, kh = lane>>4.
__device__ __forceinline__ v16bf frag_a(const bf16* p, int kh) {
  v16bf f;
#pragma unroll
  for (int j = 0; j < 8; ++j) {
    int kb = ((j >> 2) << 4) + ((j & 3) << 1) + (kh << 3);
    f[2 * j]     = p[kb];
    f[2 * j + 1] = p[kb + 1];
  }
  return f;
}

// B-matrix 32x16 bf16 fragment: lane col n = lane&15; same K order as frag_a
// (kh selects K 8..15/24..31). p = column base (column contiguous along K).
__device__ __forceinline__ v16bf frag_b(const bf16* p, int kh) {
  v16bf f;
#pragma unroll
  for (int j = 0; j < 8; ++j) {
    int kb = ((j >> 2) << 4) + ((j & 3) << 1) + (kh << 3);
    f[2 * j]     = p[kb];
    f[2 * j + 1] = p[kb + 1];
  }
  return f;
}

// ---------------- generic bf16 WMMA GEMM ----------------
// out = epi(A[M,K] @ W[K,N] + bias). BM=256 BN=64 BK=32, 256 thr (8 waves x 32 rows),
// double-buffered LDS, async A staging, pipelined B staging.
enum { EPI_BF16 = 0, EPI_F32 = 1, EPI_SIGMOID_F32 = 2, EPI_GELU_BF16 = 3, EPI_RES_F32 = 4 };

template <int EPI>
__global__ __launch_bounds__(256) void gemm_bf16_wmma(
    const bf16* __restrict__ A, const bf16* __restrict__ W,
    const float* __restrict__ bias, const float* __restrict__ res,
    void* __restrict__ out, int M, int K, int N) {
  __shared__ bf16 sA[2][256 * 32];  // [row][k]
  __shared__ bf16 sB[2][64 * 32];   // [n][k] (transposed on fill)
  const int tid  = threadIdx.x;
  const int m0   = blockIdx.x * 256;
  const int n0   = blockIdx.y * 64;
  const int wave = tid >> 5;
  const int lane = tid & 31;
  const int lm   = lane & 15;
  const int kh   = lane >> 4;

  // staging geometry
  const int arow = tid >> 2, ac8 = (tid & 3);        // + i*64 rows per step
  const int bkr  = tid >> 3, bn8 = (tid & 7) * 8;

  auto stage_a = [&](int k0, int buf) {
#pragma unroll
    for (int i = 0; i < 4; ++i) {
      int idx = tid + i * 256;
      int row = arow + i * 64;
      async_b128(&sA[buf][idx * 8],
                 A + (size_t)(m0 + row) * K + k0 + ac8 * 8);
    }
  };
  auto load_b = [&](int k0) {
    return *(const uint4*)(W + (size_t)(k0 + bkr) * N + n0 + bn8);
  };
  auto store_b = [&](uint4 raw, int buf) {
    const bf16* v = (const bf16*)&raw;
#pragma unroll
    for (int e = 0; e < 8; ++e) sB[buf][(bn8 + e) * 32 + bkr] = v[e];
  };

  v8f acc[2][4];
#pragma unroll
  for (int f = 0; f < 2; ++f)
#pragma unroll
    for (int nt = 0; nt < 4; ++nt) acc[f][nt] = (v8f)(0.f);

  const int ksteps = K >> 5;
  // prologue: tile 0
  stage_a(0, 0);
  store_b(load_b(0), 0);
  async_wait0();
  __syncthreads();

  int cur = 0;
  for (int t = 0; t < ksteps; ++t) {
    const int nxt = cur ^ 1;
    const bool more = (t + 1 < ksteps);
    uint4 braw;
    if (more) {              // issue next tile's loads before computing
      stage_a((t + 1) << 5, nxt);
      braw = load_b((t + 1) << 5);
    }
    // compute tile `cur`
    const bf16* a0 = sA[cur] + (wave * 32 + lm) * 32;
    v16bf af0 = frag_a(a0, kh);
    v16bf af1 = frag_a(a0 + 16 * 32, kh);
#pragma unroll
    for (int nt = 0; nt < 4; ++nt) {
      v16bf bfr = frag_b(sB[cur] + (nt * 16 + lm) * 32, kh);
      acc[0][nt] = WMMA_BF16(af0, bfr, acc[0][nt]);
      acc[1][nt] = WMMA_BF16(af1, bfr, acc[1][nt]);
    }
    if (more) store_b(braw, nxt);
    async_wait0();
    __syncthreads();
    cur = nxt;
  }

  // epilogue (C layout: row = wave*32 + f*16 + r + 8*kh, col = n0 + nt*16 + lm)
#pragma unroll
  for (int f = 0; f < 2; ++f) {
#pragma unroll
    for (int nt = 0; nt < 4; ++nt) {
      int col  = n0 + nt * 16 + lm;
      float bv = bias ? bias[col] : 0.0f;
#pragma unroll
      for (int r = 0; r < 8; ++r) {
        int row  = m0 + wave * 32 + f * 16 + r + kh * 8;
        float v  = acc[f][nt][r] + bv;
        size_t o = (size_t)row * N + col;
        if (EPI == EPI_BF16) {
          ((bf16*)out)[o] = (bf16)v;
        } else if (EPI == EPI_F32) {
          ((float*)out)[o] = v;
        } else if (EPI == EPI_SIGMOID_F32) {
          ((float*)out)[o] = 1.0f / (1.0f + __expf(-v));
        } else if (EPI == EPI_GELU_BF16) {
          ((bf16*)out)[o] = (bf16)(0.5f * v * (1.0f + erff(v * 0.70710678118f)));
        } else {  // EPI_RES_F32
          ((float*)out)[o] = v + res[o];
        }
      }
    }
  }
}

// ---------------- LayerNorm (f32 in, bf16 out) ----------------
__global__ __launch_bounds__(256) void layernorm_bf16(
    const float* __restrict__ x, const float* __restrict__ g,
    const float* __restrict__ b, bf16* __restrict__ out) {
  const int row = blockIdx.x;
  const float* xr = x + (size_t)row * DMODEL;
  __shared__ float red0[8], red1[8];
  float s = 0.f, s2 = 0.f;
  for (int i = threadIdx.x; i < DMODEL; i += 256) {
    float v = xr[i];
    s += v; s2 += v * v;
  }
#pragma unroll
  for (int off = 16; off; off >>= 1) {
    s  += __shfl_xor(s, off, 32);
    s2 += __shfl_xor(s2, off, 32);
  }
  if ((threadIdx.x & 31) == 0) { red0[threadIdx.x >> 5] = s; red1[threadIdx.x >> 5] = s2; }
  __syncthreads();
  s = 0.f; s2 = 0.f;
#pragma unroll
  for (int w = 0; w < 8; ++w) { s += red0[w]; s2 += red1[w]; }
  float mu   = s / (float)DMODEL;
  float var  = s2 / (float)DMODEL - mu * mu;
  float rstd = rsqrtf(var + 1e-5f);
  for (int i = threadIdx.x; i < DMODEL; i += 256) {
    float v = (xr[i] - mu) * rstd * g[i] + b[i];
    out[(size_t)row * DMODEL + i] = (bf16)v;
  }
}

// ---------------- windowed causal flash attention ----------------
// grid: (TSEQ/128, NHEAD, BSZ), 256 threads (8 waves x 16 query rows).
__global__ __launch_bounds__(256) void attn_win_wmma(
    const bf16* __restrict__ Q, const bf16* __restrict__ Kb,
    const bf16* __restrict__ V, bf16* __restrict__ O) {
  __shared__ bf16 sK[128 * 64];    // [key][dh]
  __shared__ bf16 sVt[64 * 128];   // [dh][key]
  __shared__ bf16 sP[128 * 128];   // [qrow][key]
  const int q0 = blockIdx.x * 128;
  const int h  = blockIdx.y;
  const int b  = blockIdx.z;
  const int h0 = h * DHEAD;
  const int tid = threadIdx.x, wave = tid >> 5, lane = tid & 31;
  const int lm = lane & 15, kh = lane >> 4;

  const bf16* qr = Q + ((size_t)(b * TSEQ + q0 + wave * 16 + lm)) * DMODEL + h0;
  v16bf aq0 = frag_a(qr, kh);
  v16bf aq1 = frag_a(qr + 32, kh);

  v8f oacc[4];
#pragma unroll
  for (int nt = 0; nt < 4; ++nt) oacc[nt] = (v8f)(0.f);
  float mi[8], li[8];
#pragma unroll
  for (int r = 0; r < 8; ++r) { mi[r] = -INFINITY; li[r] = 0.f; }

  for (int c = 0; c < 2; ++c) {
    const int key0 = q0 - WINSZ + c * 128;
    // stage K (async, row-major) and V (transposed via VGPR) chunk
#pragma unroll
    for (int i = 0; i < 4; ++i) {
      int idx = tid + i * 256;
      int krow = idx >> 3, c8 = idx & 7;
      int key = key0 + krow;
      uint4 vv = make_uint4(0u, 0u, 0u, 0u);
      if (key >= 0) {
        async_b128(&sK[idx * 8],
                   Kb + ((size_t)(b * TSEQ + key)) * DMODEL + h0 + c8 * 8);
        vv = *((const uint4*)(V + ((size_t)(b * TSEQ + key)) * DMODEL + h0) + c8);
      } else {
        ((uint4*)sK)[idx] = make_uint4(0u, 0u, 0u, 0u);
      }
      const bf16* ve = (const bf16*)&vv;
#pragma unroll
      for (int e = 0; e < 8; ++e) sVt[(c8 * 8 + e) * 128 + krow] = ve[e];
    }
    async_wait0();
    __syncthreads();

    // S = Q K^T over 8 key tiles
    float p[8][8];
    float rowmax[8];
#pragma unroll
    for (int r = 0; r < 8; ++r) rowmax[r] = -INFINITY;
#pragma unroll
    for (int kt = 0; kt < 8; ++kt) {
      v8f s = (v8f)(0.f);
      const bf16* kcol = sK + (kt * 16 + lm) * 64;  // n = key, k = dh
      s = WMMA_BF16(aq0, frag_b(kcol, kh), s);
      s = WMMA_BF16(aq1, frag_b(kcol + 32, kh), s);
      int keyg = key0 + kt * 16 + lm;
#pragma unroll
      for (int r = 0; r < 8; ++r) {
        int qg = q0 + wave * 16 + r + kh * 8;
        int d  = qg - keyg;
        bool ok = (keyg >= 0) && (d >= 0) && (d < WINSZ);
        float val = ok ? s[r] * 0.125f : -INFINITY;   // 1/sqrt(64)
        p[kt][r]  = val;
        rowmax[r] = fmaxf(rowmax[r], val);
      }
    }
#pragma unroll
    for (int off = 1; off < 16; off <<= 1) {
#pragma unroll
      for (int r = 0; r < 8; ++r)
        rowmax[r] = fmaxf(rowmax[r], __shfl_xor(rowmax[r], off, 32));
    }
    float alpha[8], rowsum[8];
#pragma unroll
    for (int r = 0; r < 8; ++r) {
      float mnew = fmaxf(mi[r], rowmax[r]);
      bool dead = !(mnew > -INFINITY);
      float a = dead ? 0.f : __expf(mi[r] - mnew);
      float sum = 0.f;
#pragma unroll
      for (int kt = 0; kt < 8; ++kt) {
        float e = dead ? 0.f : __expf(p[kt][r] - mnew);
        p[kt][r] = e;
        sum += e;
      }
      alpha[r] = a; rowsum[r] = sum; mi[r] = mnew;
    }
#pragma unroll
    for (int off = 1; off < 16; off <<= 1) {
#pragma unroll
      for (int r = 0; r < 8; ++r)
        rowsum[r] += __shfl_xor(rowsum[r], off, 32);
    }
#pragma unroll
    for (int r = 0; r < 8; ++r) li[r] = li[r] * alpha[r] + rowsum[r];
#pragma unroll
    for (int nt = 0; nt < 4; ++nt)
#pragma unroll
      for (int r = 0; r < 8; ++r) oacc[nt][r] *= alpha[r];

    // re-layout P (C-frag) -> LDS -> A-frag
#pragma unroll
    for (int kt = 0; kt < 8; ++kt)
#pragma unroll
      for (int r = 0; r < 8; ++r)
        sP[(wave * 16 + r + kh * 8) * 128 + kt * 16 + lm] = (bf16)p[kt][r];
    __syncthreads();

    // O += P @ V
    const bf16* prow = sP + (wave * 16 + lm) * 128;
#pragma unroll
    for (int ks = 0; ks < 4; ++ks) {
      v16bf ap = frag_a(prow + ks * 32, kh);
#pragma unroll
      for (int nt = 0; nt < 4; ++nt) {
        const bf16* vcol = sVt + (nt * 16 + lm) * 128 + ks * 32;
        oacc[nt] = WMMA_BF16(ap, frag_b(vcol, kh), oacc[nt]);
      }
    }
    __syncthreads();
  }

#pragma unroll
  for (int nt = 0; nt < 4; ++nt) {
#pragma unroll
    for (int r = 0; r < 8; ++r) {
      int qg = q0 + wave * 16 + r + kh * 8;
      float v = oacc[nt][r] / li[r];
      O[((size_t)(b * TSEQ + qg)) * DMODEL + h0 + nt * 16 + lm] = (bf16)v;
    }
  }
}

// ---------------- SSM sequential scan ----------------
__global__ void ssm_scan(const float* __restrict__ u, const float* __restrict__ Ac,
                         bf16* __restrict__ st) {
  int t0 = blockIdx.x * blockDim.x + threadIdx.x;  // 0..127
  if (t0 >= BSZ * NSSM) return;
  int b = t0 >> 6, n = t0 & 63;
  float a = Ac[n], s = 0.f;
  const float* up = u + (size_t)b * TSEQ * NSSM + n;
  bf16* sp = st + (size_t)b * TSEQ * NSSM + n;
  for (int t = 0; t < TSEQ; ++t) {
    s = a * s + up[(size_t)t * NSSM];
    sp[(size_t)t * NSSM] = (bf16)s;
  }
}

// ---------------- elementwise kernels ----------------
__global__ void cvt_f32_bf16(const float* __restrict__ in, bf16* __restrict__ out, int n) {
  for (int i = blockIdx.x * 256 + threadIdx.x; i < n; i += gridDim.x * 256)
    out[i] = (bf16)in[i];
}

__global__ void fuse_gate(const float* __restrict__ x, const float* __restrict__ g,
                          const float* __restrict__ attn, const float* __restrict__ ssm,
                          float* __restrict__ x1, int n) {
  int i = blockIdx.x * 256 + threadIdx.x;
  if (i < n) {
    float gv = g[i];
    x1[i] = x[i] + gv * attn[i] + (1.f - gv) * ssm[i];
  }
}

// ---------------- launch ----------------
static inline size_t alignup(size_t x) { return (x + 255) & ~(size_t)255; }

extern "C" void kernel_launch(void* const* d_in, const int* in_sizes, int n_in,
                              void* d_out, int out_size, void* d_ws, size_t ws_size,
                              hipStream_t stream) {
  const float* x     = (const float*)d_in[0];
  const float* ln1_g = (const float*)d_in[1];
  const float* ln1_b = (const float*)d_in[2];
  const float* WQ = (const float*)d_in[3];  const float* bQ = (const float*)d_in[4];
  const float* WK = (const float*)d_in[5];  const float* bK = (const float*)d_in[6];
  const float* WV = (const float*)d_in[7];  const float* bV = (const float*)d_in[8];
  const float* WO = (const float*)d_in[9];  const float* bO = (const float*)d_in[10];
  const float* Wg = (const float*)d_in[11]; const float* bg = (const float*)d_in[12];
  const float* Ac = (const float*)d_in[13];
  const float* Bw = (const float*)d_in[14];
  const float* Cw = (const float*)d_in[15];
  const float* ln2_g = (const float*)d_in[16];
  const float* ln2_b = (const float*)d_in[17];
  const float* W1 = (const float*)d_in[18]; const float* b1 = (const float*)d_in[19];
  const float* W2 = (const float*)d_in[20]; const float* b2 = (const float*)d_in[21];
  float* out = (float*)d_out;

  char* w = (char*)d_ws;
  auto alloc = [&](size_t bytes) { char* p = w; w += alignup(bytes); return p; };

  bf16* wq_bf = (bf16*)alloc((size_t)DMODEL * DMODEL * 2);
  bf16* wk_bf = (bf16*)alloc((size_t)DMODEL * DMODEL * 2);
  bf16* wv_bf = (bf16*)alloc((size_t)DMODEL * DMODEL * 2);
  bf16* wo_bf = (bf16*)alloc((size_t)DMODEL * DMODEL * 2);
  bf16* wg_bf = (bf16*)alloc((size_t)DMODEL * DMODEL * 2);
  bf16* w1_bf = (bf16*)alloc((size_t)DMODEL * DFFN * 2);
  bf16* w2_bf = (bf16*)alloc((size_t)DFFN * DMODEL * 2);
  bf16* bw_bf = (bf16*)alloc((size_t)DMODEL * NSSM * 2);
  bf16* cw_bf = (bf16*)alloc((size_t)NSSM * DMODEL * 2);
  bf16* h_bf  = (bf16*)alloc((size_t)MROWS * DMODEL * 2);
  bf16* q_bf  = (bf16*)alloc((size_t)MROWS * DMODEL * 2);
  bf16* k_bf  = (bf16*)alloc((size_t)MROWS * DMODEL * 2);
  bf16* v_bf  = (bf16*)alloc((size_t)MROWS * DMODEL * 2);
  bf16* ao_bf = (bf16*)alloc((size_t)MROWS * DMODEL * 2);
  bf16* st_bf = (bf16*)alloc((size_t)MROWS * NSSM * 2);
  bf16* h2_bf = (bf16*)alloc((size_t)MROWS * DMODEL * 2);
  bf16* mid_bf= (bf16*)alloc((size_t)MROWS * DFFN * 2);
  float* gsig = (float*)alloc((size_t)MROWS * DMODEL * 4);
  float* attn = (float*)alloc((size_t)MROWS * DMODEL * 4);
  float* ufp  = (float*)alloc((size_t)MROWS * NSSM * 4);
  float* ssm  = (float*)alloc((size_t)MROWS * DMODEL * 4);
  float* x1   = (float*)alloc((size_t)MROWS * DMODEL * 4);

  auto cvt = [&](const float* src, bf16* dst, int n) {
    int blocks = (n + 255) / 256; if (blocks > 4096) blocks = 4096;
    hipLaunchKernelGGL(cvt_f32_bf16, dim3(blocks), dim3(256), 0, stream, src, dst, n);
  };
  cvt(WQ, wq_bf, DMODEL * DMODEL); cvt(WK, wk_bf, DMODEL * DMODEL);
  cvt(WV, wv_bf, DMODEL * DMODEL); cvt(WO, wo_bf, DMODEL * DMODEL);
  cvt(Wg, wg_bf, DMODEL * DMODEL);
  cvt(W1, w1_bf, DMODEL * DFFN);   cvt(W2, w2_bf, DFFN * DMODEL);
  cvt(Bw, bw_bf, DMODEL * NSSM);   cvt(Cw, cw_bf, NSSM * DMODEL);

  hipLaunchKernelGGL(layernorm_bf16, dim3(MROWS), dim3(256), 0, stream, x, ln1_g, ln1_b, h_bf);

  dim3 gD(MROWS / 256, DMODEL / 64);
  hipLaunchKernelGGL((gemm_bf16_wmma<EPI_BF16>), gD, dim3(256), 0, stream,
                     h_bf, wq_bf, bQ, (const float*)nullptr, (void*)q_bf, MROWS, DMODEL, DMODEL);
  hipLaunchKernelGGL((gemm_bf16_wmma<EPI_BF16>), gD, dim3(256), 0, stream,
                     h_bf, wk_bf, bK, (const float*)nullptr, (void*)k_bf, MROWS, DMODEL, DMODEL);
  hipLaunchKernelGGL((gemm_bf16_wmma<EPI_BF16>), gD, dim3(256), 0, stream,
                     h_bf, wv_bf, bV, (const float*)nullptr, (void*)v_bf, MROWS, DMODEL, DMODEL);
  hipLaunchKernelGGL((gemm_bf16_wmma<EPI_SIGMOID_F32>), gD, dim3(256), 0, stream,
                     h_bf, wg_bf, bg, (const float*)nullptr, (void*)gsig, MROWS, DMODEL, DMODEL);
  hipLaunchKernelGGL((gemm_bf16_wmma<EPI_F32>), dim3(MROWS / 256, NSSM / 64), dim3(256), 0, stream,
                     h_bf, bw_bf, (const float*)nullptr, (const float*)nullptr,
                     (void*)ufp, MROWS, DMODEL, NSSM);

  hipLaunchKernelGGL(attn_win_wmma, dim3(TSEQ / 128, NHEAD, BSZ), dim3(256), 0, stream,
                     q_bf, k_bf, v_bf, ao_bf);

  hipLaunchKernelGGL((gemm_bf16_wmma<EPI_F32>), gD, dim3(256), 0, stream,
                     ao_bf, wo_bf, bO, (const float*)nullptr, (void*)attn, MROWS, DMODEL, DMODEL);

  hipLaunchKernelGGL(ssm_scan, dim3(1), dim3(128), 0, stream, ufp, Ac, st_bf);
  hipLaunchKernelGGL((gemm_bf16_wmma<EPI_F32>), gD, dim3(256), 0, stream,
                     st_bf, cw_bf, (const float*)nullptr, (const float*)nullptr,
                     (void*)ssm, MROWS, NSSM, DMODEL);

  int total = MROWS * DMODEL;
  hipLaunchKernelGGL(fuse_gate, dim3((total + 255) / 256), dim3(256), 0, stream,
                     x, gsig, attn, ssm, x1, total);

  hipLaunchKernelGGL(layernorm_bf16, dim3(MROWS), dim3(256), 0, stream, x1, ln2_g, ln2_b, h2_bf);
  hipLaunchKernelGGL((gemm_bf16_wmma<EPI_GELU_BF16>), dim3(MROWS / 256, DFFN / 64), dim3(256), 0, stream,
                     h2_bf, w1_bf, b1, (const float*)nullptr, (void*)mid_bf, MROWS, DMODEL, DFFN);
  hipLaunchKernelGGL((gemm_bf16_wmma<EPI_RES_F32>), gD, dim3(256), 0, stream,
                     mid_bf, w2_bf, b2, x1, (void*)out, MROWS, DFFN, DMODEL);

  (void)in_sizes; (void)n_in; (void)out_size; (void)ws_size;
}
